// WindowAttention_64355789964113
// MI455X (gfx1250) — matrix operand order repaired and checked by reading, not compiled, not run on hardware
//
#include <hip/hip_runtime.h>
#include <hip/hip_bf16.h>

typedef __attribute__((ext_vector_type(16))) __bf16 v16bf;
typedef __attribute__((ext_vector_type(8)))  __bf16 v8bf;
typedef __attribute__((ext_vector_type(8)))  float  v8f;

#define NUM_DIM   384
#define NUM_HEADS 12
#define HEAD_DIM  32
#define WIN_L     49
#define LPAD      64
#define NB        1024
#define MROWS     (NB*WIN_L)    // 50176 = 16*3136
#define MTILES    (MROWS/16)    // 3136
#define MBLOCKS   (MTILES/8)    // 392 (8 m-tiles per 256-thread block)
#define QKV_N     (3*NUM_DIM)   // 1152
#define KSTEPS    (NUM_DIM/32)  // 12

__device__ __forceinline__ v8f wmma_bf16(v16bf a, v16bf b, v8f c) {
  // 8 args: (neg_a, A, neg_b, B, c_mod, C, reuse_a, reuse_b)
  return __builtin_amdgcn_wmma_f32_16x16x32_bf16(false, a, false, b, (short)0, c, false, false);
}

__device__ __forceinline__ v16bf cat16(v8bf lo, v8bf hi) {
  v16bf r;
#pragma unroll
  for (int i = 0; i < 8; ++i) { r[i] = lo[i]; r[i + 8] = hi[i]; }
  return r;
}

__device__ __forceinline__ v16bf zero16() {
  v16bf r;
#pragma unroll
  for (int i = 0; i < 16; ++i) r[i] = (__bf16)0.0f;
  return r;
}

__device__ __forceinline__ v8f zero8() {
  v8f r;
#pragma unroll
  for (int i = 0; i < 8; ++i) r[i] = 0.0f;
  return r;
}

// ---------------------------------------------------------------------------
// Kernel 0: one-time fp32 -> bf16 weight conversion (weights reused 50176x)
// ---------------------------------------------------------------------------
__global__ void k_convert_w(const float* __restrict__ qkv_w,
                            const float* __restrict__ proj_w,
                            __bf16* __restrict__ wqkv,
                            __bf16* __restrict__ wproj) {
  int i = blockIdx.x * 256 + threadIdx.x;
  if (i < QKV_N * NUM_DIM)   wqkv[i]  = (__bf16)qkv_w[i];
  if (i < NUM_DIM * NUM_DIM) wproj[i] = (__bf16)proj_w[i];
}

// ---------------------------------------------------------------------------
// Kernel 1: QKV GEMM (50176 x 384) @ (384 x 1152) -> q, k, v^T (bf16).
// Block = 8 waves = 128x64 output tile. B k-slab (64 cols x 32 k = 4 KB)
// double-buffered in LDS, shared by all 8 waves (8x cut in L2 B-traffic).
// ---------------------------------------------------------------------------
__global__ __launch_bounds__(256) void k_qkv(const float* __restrict__ x,
                                             const __bf16* __restrict__ wqkv,
                                             __bf16* __restrict__ qbuf,
                                             __bf16* __restrict__ kbuf,
                                             __bf16* __restrict__ vT) {
  __shared__ alignas(32) __bf16 s_b[2][64 * 32];   // 8 KB double buffer

  const int lane = threadIdx.x & 31;
  const int wave = threadIdx.x >> 5;
  const int half = lane >> 4;
  const int l16  = lane & 15;
  const int mblk = blockIdx.x / 18;
  const int ng   = blockIdx.x - mblk * 18;         // 18 groups of 64 out cols
  const int mt   = mblk * 8 + wave;
  const int mbase = mt * 16;

  // Cooperative B staging assignment: 256 threads cover 64 rows x 4 chunks of 8
  const int srow   = threadIdx.x >> 2;             // 0..63
  const int schunk = (threadIdx.x & 3) * 8;        // 0,8,16,24
  const __bf16* bsrc = wqkv + (size_t)(ng * 64 + srow) * NUM_DIM + schunk;
  __bf16* bdst0 = &s_b[0][srow * 32 + schunk];
  __bf16* bdst1 = &s_b[1][srow * 32 + schunk];

  v8f acc[4];
#pragma unroll
  for (int i = 0; i < 4; ++i) acc[i] = zero8();

  const float* xrow = x + (size_t)(mbase + l16) * NUM_DIM;

  // preload k-step 0
  *(v8bf*)bdst0 = *(const v8bf*)(bsrc);
  __syncthreads();

  for (int kt = 0; kt < KSTEPS; ++kt) {
    const int cur = kt & 1;
    // prefetch next k-slab into the other buffer
    if (kt + 1 < KSTEPS) {
      __bf16* d = cur ? bdst0 : bdst1;
      *(v8bf*)d = *(const v8bf*)(bsrc + (kt + 1) * 32);
    }
    // A fragment: row (mbase+l16), K-chunk [half*16 .. +15]
    const float* pa = xrow + kt * 32 + half * 16;
    v8bf lo, hi;
#pragma unroll
    for (int i = 0; i < 8; ++i) { lo[i] = (__bf16)pa[i]; hi[i] = (__bf16)pa[8 + i]; }
    v16bf a = cat16(lo, hi);
#pragma unroll
    for (int nt = 0; nt < 4; ++nt) {
      v16bf b = *(const v16bf*)(&s_b[cur][(nt * 16 + l16) * 32 + half * 16]);
      acc[nt] = wmma_bf16(a, b, acc[nt]);
    }
    __syncthreads();
  }

  // Scatter: r = which*384 + h*32 + d  (q,k: [b,h,l,d]; v: transposed [b,h,d,64])
#pragma unroll
  for (int nt = 0; nt < 4; ++nt) {
    int rout  = ng * 64 + nt * 16 + l16;
    int which = rout / NUM_DIM;
    int rem   = rout - which * NUM_DIM;
    int h     = rem >> 5;
    int d     = rem & 31;
#pragma unroll
    for (int r = 0; r < 8; ++r) {
      int g    = mbase + r + 8 * half;             // global row < 50176
      int b_   = g / WIN_L;
      int lrow = g - b_ * WIN_L;
      __bf16 bv = (__bf16)acc[nt][r];
      size_t bh = (size_t)b_ * NUM_HEADS + h;
      if (which == 0)      qbuf[(bh * WIN_L + lrow) * HEAD_DIM + d] = bv;
      else if (which == 1) kbuf[(bh * WIN_L + lrow) * HEAD_DIM + d] = bv;
      else                 vT[(bh * HEAD_DIM + d) * LPAD + lrow]    = bv;
    }
  }
}

// ---------------------------------------------------------------------------
// Kernel 2: fused attention per (b, h). 128 threads = 4 waves.
// logits = (q k^T)*scale + bias, mask -> -100, softmax, out = P @ v
// ---------------------------------------------------------------------------
__global__ __launch_bounds__(128) void k_attn(const __bf16* __restrict__ qbuf,
                                              const __bf16* __restrict__ kbuf,
                                              const __bf16* __restrict__ vT,
                                              const unsigned char* __restrict__ mask,
                                              const float* __restrict__ rel_bias,
                                              const int* __restrict__ rel_coords,
                                              __bf16* __restrict__ y) {
  __shared__ float  s_logit[LPAD * LPAD];          // 16 KB
  __shared__ alignas(32) __bf16 s_p[LPAD * LPAD];  //  8 KB

  const int bh   = blockIdx.x;
  const int b_   = bh / NUM_HEADS;
  const int h    = bh - b_ * NUM_HEADS;
  const int lane = threadIdx.x & 31;
  const int wave = threadIdx.x >> 5;
  const int half = lane >> 4;
  const int l16  = lane & 15;
  const size_t qkbase = (size_t)bh * WIN_L * HEAD_DIM;
  const float scale = 19.595917942265423f;         // sqrt(384), per reference

  // ---- Phase A: logits = q @ k^T (M=64 pad, N=64 pad, K=32) ----
  {
    const int mt = wave;
    const int m  = mt * 16 + l16;
    v16bf a = zero16();
    if (m < WIN_L) {
      const __bf16* p = qbuf + qkbase + (size_t)m * HEAD_DIM + half * 16;
      a = cat16(*(const v8bf*)p, *(const v8bf*)(p + 8));
    }
    const unsigned char* mrow = mask + (size_t)(b_ & 3) * WIN_L * WIN_L;
#pragma unroll
    for (int nt = 0; nt < 4; ++nt) {
      const int n = nt * 16 + l16;
      v16bf bb = zero16();
      if (n < WIN_L)
        bb = *(const v16bf*)(kbuf + qkbase + (size_t)n * HEAD_DIM + half * 16);
      v8f d = wmma_bf16(a, bb, zero8());
#pragma unroll
      for (int r = 0; r < 8; ++r) {
        int mm = mt * 16 + r + 8 * half;
        float val = -1e30f;
        if (mm < WIN_L && n < WIN_L) {
          val = d[r] * scale +
                rel_bias[(size_t)rel_coords[mm * WIN_L + n] * NUM_HEADS + h];
          if (mrow[mm * WIN_L + n]) val = -100.0f;
        }
        s_logit[mm * LPAD + n] = val;
      }
    }
  }
  __syncthreads();

  // ---- Phase B: row softmax over the 49 valid columns ----
  {
    const int m = threadIdx.x;
    if (m < LPAD) {
      if (m < WIN_L) {
        float mx = -1e30f;
        for (int n = 0; n < WIN_L; ++n) mx = fmaxf(mx, s_logit[m * LPAD + n]);
        float sum = 0.0f;
        for (int n = 0; n < WIN_L; ++n) sum += __expf(s_logit[m * LPAD + n] - mx);
        float inv = 1.0f / sum;
        for (int n = 0; n < WIN_L; ++n)
          s_p[m * LPAD + n] = (__bf16)(__expf(s_logit[m * LPAD + n] - mx) * inv);
        for (int n = WIN_L; n < LPAD; ++n) s_p[m * LPAD + n] = (__bf16)0.0f;
      } else {
        for (int n = 0; n < LPAD; ++n) s_p[m * LPAD + n] = (__bf16)0.0f;
      }
    }
  }
  __syncthreads();

  // ---- Phase C: out = P @ v  (M=64 pad, N=32, K=64 pad; pad probs are 0) ----
  {
    const int mt = wave;
    const int m  = mt * 16 + l16;
    const size_t vbase = (size_t)bh * HEAD_DIM * LPAD;
#pragma unroll
    for (int nt = 0; nt < 2; ++nt) {
      v8f acc = zero8();
#pragma unroll
      for (int ks = 0; ks < 2; ++ks) {
        const __bf16* p = s_p + m * LPAD + ks * 32 + half * 16;
        v16bf a = cat16(*(const v8bf*)p, *(const v8bf*)(p + 8));
        int n = nt * 16 + l16;   // output channel d
        v16bf bb = *(const v16bf*)(vT + vbase + (size_t)n * LPAD + ks * 32 + half * 16);
        acc = wmma_bf16(a, bb, acc);
      }
#pragma unroll
      for (int r = 0; r < 8; ++r) {
        int mm = mt * 16 + r + 8 * half;
        if (mm < WIN_L) {
          int n = nt * 16 + l16;
          y[((size_t)b_ * WIN_L + mm) * NUM_DIM + h * HEAD_DIM + n] = (__bf16)acc[r];
        }
      }
    }
  }
}

// ---------------------------------------------------------------------------
// Kernel 3: projection GEMM (50176 x 384) @ (384 x 384) + bias -> f32 out.
// Same block tiling / LDS B staging as kernel 1.
// ---------------------------------------------------------------------------
__global__ __launch_bounds__(256) void k_proj(const __bf16* __restrict__ y,
                                              const __bf16* __restrict__ wproj,
                                              const float* __restrict__ proj_b,
                                              float* __restrict__ out) {
  __shared__ alignas(32) __bf16 s_b[2][64 * 32];   // 8 KB double buffer

  const int lane = threadIdx.x & 31;
  const int wave = threadIdx.x >> 5;
  const int half = lane >> 4;
  const int l16  = lane & 15;
  const int mblk = blockIdx.x / 6;
  const int ng   = blockIdx.x - mblk * 6;
  const int mt   = mblk * 8 + wave;
  const int mbase = mt * 16;

  const int srow   = threadIdx.x >> 2;
  const int schunk = (threadIdx.x & 3) * 8;
  const __bf16* bsrc = wproj + (size_t)(ng * 64 + srow) * NUM_DIM + schunk;
  __bf16* bdst0 = &s_b[0][srow * 32 + schunk];
  __bf16* bdst1 = &s_b[1][srow * 32 + schunk];

  v8f acc[4];
#pragma unroll
  for (int i = 0; i < 4; ++i) acc[i] = zero8();

  const __bf16* yrow = y + (size_t)(mbase + l16) * NUM_DIM;

  *(v8bf*)bdst0 = *(const v8bf*)(bsrc);
  __syncthreads();

  for (int kt = 0; kt < KSTEPS; ++kt) {
    const int cur = kt & 1;
    if (kt + 1 < KSTEPS) {
      __bf16* d = cur ? bdst0 : bdst1;
      *(v8bf*)d = *(const v8bf*)(bsrc + (kt + 1) * 32);
    }
    const __bf16* pa = yrow + kt * 32 + half * 16;
    v16bf a = cat16(*(const v8bf*)pa, *(const v8bf*)(pa + 8));
#pragma unroll
    for (int nt = 0; nt < 4; ++nt) {
      v16bf b = *(const v16bf*)(&s_b[cur][(nt * 16 + l16) * 32 + half * 16]);
      acc[nt] = wmma_bf16(a, b, acc[nt]);
    }
    __syncthreads();
  }

#pragma unroll
  for (int nt = 0; nt < 4; ++nt) {
    int n = ng * 64 + nt * 16 + l16;
    float bias = proj_b[n];
#pragma unroll
    for (int r = 0; r < 8; ++r) {
      int g = mbase + r + 8 * half;
      out[(size_t)g * NUM_DIM + n] = acc[nt][r] + bias;
    }
  }
}

// ---------------------------------------------------------------------------
extern "C" void kernel_launch(void* const* d_in, const int* in_sizes, int n_in,
                              void* d_out, int out_size, void* d_ws, size_t ws_size,
                              hipStream_t stream) {
  const float*         x          = (const float*)d_in[0];
  const unsigned char* mask       = (const unsigned char*)d_in[1];  // jax bool
  const float*         qkv_w      = (const float*)d_in[2];
  const float*         proj_w     = (const float*)d_in[3];
  const float*         proj_b     = (const float*)d_in[4];
  const float*         rel_bias   = (const float*)d_in[5];
  const int*           rel_coords = (const int*)d_in[6];
  float*               out        = (float*)d_out;

  char* ws = (char*)d_ws;
  size_t off = 0;
  auto alloc = [&](size_t bytes) -> char* {
    char* p = ws + off;
    off += (bytes + 255) & ~(size_t)255;
    return p;
  };
  __bf16* wqkv  = (__bf16*)alloc((size_t)QKV_N * NUM_DIM * 2);                   // ~0.9 MB
  __bf16* wproj = (__bf16*)alloc((size_t)NUM_DIM * NUM_DIM * 2);                 // ~0.3 MB
  __bf16* qbuf  = (__bf16*)alloc((size_t)NB * NUM_HEADS * WIN_L * HEAD_DIM * 2); // 38.5 MB
  __bf16* kbuf  = (__bf16*)alloc((size_t)NB * NUM_HEADS * WIN_L * HEAD_DIM * 2); // 38.5 MB
  size_t vT_bytes = (size_t)NB * NUM_HEADS * HEAD_DIM * LPAD * 2;                // 50.3 MB
  __bf16* vT    = (__bf16*)alloc(vT_bytes);
  __bf16* y     = (__bf16*)alloc((size_t)MROWS * NUM_DIM * 2);                   // 38.5 MB

  // Zero v^T so the L-pad region (cols 49..63) is 0 (paired with zero probs).
  hipMemsetAsync(vT, 0, vT_bytes, stream);

  k_convert_w<<<(QKV_N * NUM_DIM + 255) / 256, 256, 0, stream>>>(qkv_w, proj_w, wqkv, wproj);
  k_qkv<<<MBLOCKS * 18, 256, 0, stream>>>(x, wqkv, qbuf, kbuf, vT);
  k_attn<<<NB * NUM_HEADS, 128, 0, stream>>>(qbuf, kbuf, vT, mask, rel_bias, rel_coords, y);
  k_proj<<<MBLOCKS * 6, 256, 0, stream>>>(y, wproj, proj_b, out);
}
